// MixedAttnHeadEmbed_26164940767960
// MI455X (gfx1250) — matrix operand order repaired and checked by reading, not compiled, not run on hardware
//
#include <hip/hip_runtime.h>

// ---------------- common types ----------------
typedef __attribute__((ext_vector_type(16))) __bf16 v16bf;
typedef __attribute__((ext_vector_type(8)))  float  v8f;

union Frag16 {
    v16bf v;
    uint4 u[2];
    unsigned short s[16];
};

static __device__ __forceinline__ unsigned short f2bf(float f) {
    unsigned int u = __builtin_bit_cast(unsigned int, f);
    u += 0x7FFFu + ((u >> 16) & 1u);          // round-to-nearest-even
    return (unsigned short)(u >> 16);
}

static __device__ __forceinline__ int imin(int a, int b) { return a < b ? a : b; }

// ---------------- problem constants ----------------
#define BB   4
#define TT   1024
#define MH   2048
#define NEGV (-1.0e9f)

// workspace layout (bf16 elements)
#define QOFF0 ((size_t)0)          // B*T*2048
#define QOFF1 ((size_t)8388608)
#define KOFF0 ((size_t)16777216)   // B*8*T*256
#define KOFF1 ((size_t)25165824)   // B*8*T*128
#define VOFF0 ((size_t)29360128)   // transposed [b][kvh][D][T]
#define VOFF1 ((size_t)37748736)

// ============================================================
// Prep: RoPE + weighted mix + pad, f32 -> bf16, WMMA-friendly layouts
// one block per (b, t); 256 threads
// ============================================================
__global__ __launch_bounds__(256) void mixprep_kernel(
    const float* __restrict__ qm, const float* __restrict__ km,
    const float* __restrict__ vm, const float* __restrict__ wts,
    unsigned short* __restrict__ ws)
{
    const int bt  = blockIdx.x;
    const int b   = bt >> 10;
    const int t   = bt & 1023;
    const int tid = threadIdx.x;

    const float w[4] = { wts[0], wts[1], wts[2], wts[3] };
    const float* __restrict__ qrow = qm + (size_t)bt * MH;
    const float* __restrict__ krow = km + (size_t)bt * MH;
    const float* __restrict__ vrow = vm + (size_t)bt * MH;

    #pragma unroll
    for (int cfg = 0; cfg < 2; ++cfg) {
        const int H  = 8 << cfg;          // 8, 16
        const int Dm = MH / H;            // 256, 128
        unsigned short* Qw = ws + (cfg ? QOFF1 : QOFF0);
        unsigned short* Kw = ws + (cfg ? KOFF1 : KOFF0);
        unsigned short* Vw = ws + (cfg ? VOFF1 : VOFF0);

        // ---- Q (H heads) ----
        for (int idx = tid; idx < MH; idx += 256) {
            const int hh = idx / Dm, j = idx - hh * Dm;
            float acc = 0.f;
            #pragma unroll
            for (int ei = 0; ei < 2; ++ei) {
                const int d = (1024 << ei) / H;
                if (j < d) {
                    const int half = d >> 1;
                    const int i    = (j < half) ? j : j - half;
                    const float ang = (float)t *
                        __expf(-9.210340372f * (2.f * (float)i) / (float)d);
                    float sn, cs; __sincosf(ang, &sn, &cs);
                    const float x  = qrow[hh * d + j];
                    const float xr = (j < half) ? -qrow[hh * d + j + half]
                                                :  qrow[hh * d + j - half];
                    acc += w[cfg * 2 + ei] * (x * cs + xr * sn);
                }
            }
            Qw[(((size_t)b * H + hh) * TT + t) * Dm + j] = f2bf(acc);
        }
        // ---- K (8 kv heads, RoPE) ----
        for (int idx = tid; idx < 8 * Dm; idx += 256) {
            const int hh = idx / Dm, j = idx - hh * Dm;
            float acc = 0.f;
            #pragma unroll
            for (int ei = 0; ei < 2; ++ei) {
                const int d = (1024 << ei) / H;
                if (j < d) {
                    const int half = d >> 1;
                    const int i    = (j < half) ? j : j - half;
                    const float ang = (float)t *
                        __expf(-9.210340372f * (2.f * (float)i) / (float)d);
                    float sn, cs; __sincosf(ang, &sn, &cs);
                    const float x  = krow[hh * d + j];
                    const float xr = (j < half) ? -krow[hh * d + j + half]
                                                :  krow[hh * d + j - half];
                    acc += w[cfg * 2 + ei] * (x * cs + xr * sn);
                }
            }
            Kw[(((size_t)b * 8 + hh) * TT + t) * Dm + j] = f2bf(acc);
        }
        // ---- V (8 kv heads, no RoPE) -> transposed [b][kvh][D][T] ----
        for (int idx = tid; idx < 8 * Dm; idx += 256) {
            const int hh = idx / Dm, j = idx - hh * Dm;
            float acc = 0.f;
            #pragma unroll
            for (int ei = 0; ei < 2; ++ei) {
                const int d = (1024 << ei) / H;
                if (j < d) acc += w[cfg * 2 + ei] * vrow[hh * d + j];
            }
            Vw[(((size_t)b * 8 + hh) * Dm + j) * TT + t] = f2bf(acc);
        }
    }
}

// ============================================================
// Flash attention, wave32 WMMA bf16. 4 waves/block, 16 q-rows/wave.
// Block-uniform key-chunk loop: K chunk staged once per block into LDS
// via GLOBAL_LOAD_ASYNC_TO_LDS_B128 (ASYNCcnt), shared by all 4 waves.
// Q A-fragments resident in registers. QK^T inner loop is software-
// pipelined (2 frag buffers) with dual accumulators to overlap ds_load
// with the matrix pipe and halve the WMMA RAW chain.
// ============================================================
template<int H, int D>
__global__ __launch_bounds__(128) void flashattn_kernel(
    const unsigned short* __restrict__ Q,
    const unsigned short* __restrict__ K,
    const unsigned short* __restrict__ Vt,
    float* __restrict__ out)
{
    constexpr int KC   = 64;      // keys per staged chunk
    constexpr int KPAD = 8;       // bf16 elems of row padding (bank spread)
    constexpr int NREP = H / 8;
    constexpr int NDT  = D / 16;

    __shared__ float          S[4][16][KC + 4];      // per-wave score chunk
    __shared__ unsigned short Kt[KC][D + KPAD];      // block-shared K chunk

    const int lane = threadIdx.x & 31;
    const int wv   = threadIdx.x >> 5;
    const int lm   = lane & 15;
    const int hi   = (lane >> 4) & 1;

    int bid = blockIdx.x;
    const int QT = TT / 64;
    const int qt = bid % QT;  bid /= QT;
    const int h  = bid % H;
    const int b  = bid / H;
    const int kvh = h / NREP;
    const int q0  = qt * 64 + wv * 16;
    const int kend     = q0 + 16;       // this wave's causal end
    const int kend_blk = qt * 64 + 64;  // block-uniform causal end
    const float scale = rsqrtf((float)D);

    const unsigned short* __restrict__ Qrow =
        Q + (((size_t)b * H + h) * TT + (size_t)(q0 + lm)) * D;
    const unsigned short* __restrict__ Kbase =
        K + ((size_t)b * 8 + kvh) * TT * (size_t)D;
    const unsigned short* __restrict__ Vbase =
        Vt + ((size_t)b * 8 + kvh) * (size_t)D * TT;

    // ---- preload this wave's Q A-fragments (invariant across key loop) ----
    Frag16 qf[D / 32];
    #pragma unroll
    for (int d0 = 0; d0 < D; d0 += 32) {
        const unsigned short* qp = Qrow + d0 + hi * 16;
        qf[d0 / 32].u[0] = *(const uint4*)(qp);        // K = b16+0..7
        qf[d0 / 32].u[1] = *(const uint4*)(qp + 8);    // K = b16+8..15
    }

    v8f O[NDT];
    #pragma unroll
    for (int dt = 0; dt < NDT; ++dt)
        O[dt] = (v8f){0.f,0.f,0.f,0.f,0.f,0.f,0.f,0.f};
    float m_run = -1.0e30f, l_run = 0.f;

    const unsigned ldsK = (unsigned)(uintptr_t)(&Kt[0][0]);
    const unsigned long long kgbase = (unsigned long long)(uintptr_t)Kbase;

    for (int kc0 = 0; kc0 < kend_blk; kc0 += KC) {
        // ---- cooperative async stage of K chunk: global -> LDS ----
        for (int i = threadIdx.x; i < KC * (D / 8); i += 128) {
            const int row = i / (D / 8);
            const int c8  = i - row * (D / 8);
            const unsigned goff = (unsigned)(((kc0 + row) * D + c8 * 8) * 2);
            const unsigned loff = ldsK + (unsigned)((row * (D + KPAD) + c8 * 8) * 2);
            asm volatile("global_load_async_to_lds_b128 %0, %1, %2"
                         :: "v"(loff), "v"(goff), "s"(kgbase) : "memory");
        }
        asm volatile("s_wait_asynccnt 0" ::: "memory");
        __syncthreads();

        if (kc0 < kend) {
            const int clen = imin(KC, kend - kc0);

            // ---- S = (Q K^T) * scale, causal-masked, into LDS ----
            for (int nt = 0; nt < clen / 16; ++nt) {
                const int kl = nt * 16 + lm;       // key within chunk (N col)
                const int kg = kc0 + kl;           // global key
                const unsigned short* kp0 = &Kt[kl][hi * 16];

                v8f acc0 = (v8f){0.f,0.f,0.f,0.f,0.f,0.f,0.f,0.f};
                v8f acc1 = (v8f){0.f,0.f,0.f,0.f,0.f,0.f,0.f,0.f};
                Frag16 fb0, fb1;
                fb0.u[0] = *(const uint4*)(kp0);
                fb0.u[1] = *(const uint4*)(kp0 + 8);
                #pragma unroll
                for (int d0 = 0; d0 < D; d0 += 64) {
                    fb1.u[0] = *(const uint4*)(kp0 + d0 + 32);
                    fb1.u[1] = *(const uint4*)(kp0 + d0 + 40);
                    acc0 = __builtin_amdgcn_wmma_f32_16x16x32_bf16(
                            false, qf[d0 / 32].v, false, fb0.v, (short)0, acc0,
                            false, false);
                    if (d0 + 64 < D) {
                        fb0.u[0] = *(const uint4*)(kp0 + d0 + 64);
                        fb0.u[1] = *(const uint4*)(kp0 + d0 + 72);
                    }
                    acc1 = __builtin_amdgcn_wmma_f32_16x16x32_bf16(
                            false, qf[d0 / 32 + 1].v, false, fb1.v, (short)0, acc1,
                            false, false);
                }
                const v8f acc = acc0 + acc1;
                #pragma unroll
                for (int vr = 0; vr < 8; ++vr) {
                    const int M = vr + hi * 8;
                    const float sv = (kg <= q0 + M) ? acc[vr] * scale : NEGV;
                    S[wv][M][nt * 16 + lm] = sv;
                }
            }
            asm volatile("s_wait_dscnt 0" ::: "memory");

            // ---- online softmax (lane lm owns row lm; halves split cols) ----
            float* __restrict__ prow = &S[wv][lm][0];
            const int c_lo = hi * 32, c_hi = imin(hi * 32 + 32, clen);
            float mx = -1.0e30f;
            for (int c = c_lo; c < c_hi; ++c) mx = fmaxf(mx, prow[c]);
            mx = fmaxf(mx, __shfl_xor(mx, 16, 32));
            const float m_new = fmaxf(m_run, mx);
            const float esc   = __expf(m_run - m_new);
            float sm = 0.f;
            for (int c = c_lo; c < c_hi; ++c) {
                const float p = __expf(prow[c] - m_new);
                prow[c] = p;
                sm += p;
            }
            sm += __shfl_xor(sm, 16, 32);
            l_run = l_run * esc + sm;
            m_run = m_new;

            // rescale running O (per-row factors via lane permute)
            float fac8[8];
            #pragma unroll
            for (int vr = 0; vr < 8; ++vr) fac8[vr] = __shfl(esc, vr + hi * 8, 32);
            #pragma unroll
            for (int dt = 0; dt < NDT; ++dt)
                #pragma unroll
                for (int vr = 0; vr < 8; ++vr) O[dt][vr] *= fac8[vr];

            // zero-pad P to a multiple of 32 keys for the WMMA sweep
            if (clen & 31) {
                if (lane < 16) {
                    #pragma unroll
                    for (int i = 0; i < 16; ++i) S[wv][lane][clen + i] = 0.f;
                }
            }
            asm volatile("s_wait_dscnt 0" ::: "memory");

            // ---- O += P * V  (V direct from transposed global layout) ----
            for (int kb = 0; kb < clen; kb += 32) {
                Frag16 ap;
                const float* __restrict__ pr = &S[wv][lm][kb + hi * 8];
                #pragma unroll
                for (int i = 0; i < 8; ++i) {
                    ap.s[i]     = f2bf(pr[i]);        // kk = b8+0..7
                    ap.s[i + 8] = f2bf(pr[16 + i]);   // kk = b8+16..23
                }
                #pragma unroll
                for (int dt = 0; dt < NDT; ++dt) {
                    Frag16 bv;
                    const unsigned short* vp =
                        Vbase + (size_t)(dt * 16 + lm) * TT + kc0 + kb + hi * 8;
                    bv.u[0] = *(const uint4*)(vp);
                    bv.u[1] = *(const uint4*)(vp + 16);
                    O[dt] = __builtin_amdgcn_wmma_f32_16x16x32_bf16(
                              false, ap.v, false, bv.v, (short)0, O[dt],
                              false, false);
                }
            }
        }
        __syncthreads();   // protect Kt before next chunk refill
    }

    // ---- normalize and accumulate into output ----
    const float inv = 1.f / l_run;
    float fac8[8];
    #pragma unroll
    for (int vr = 0; vr < 8; ++vr) fac8[vr] = __shfl(inv, vr + hi * 8, 32);
    #pragma unroll
    for (int dt = 0; dt < NDT; ++dt) {
        #pragma unroll
        for (int vr = 0; vr < 8; ++vr) {
            const int row = q0 + vr + hi * 8;
            const int col = h * D + dt * 16 + lm;
            atomicAdd(&out[((size_t)b * TT + row) * MH + col], O[dt][vr] * fac8[vr]);
        }
    }
}

// ============================================================
extern "C" void kernel_launch(void* const* d_in, const int* in_sizes, int n_in,
                              void* d_out, int out_size, void* d_ws, size_t ws_size,
                              hipStream_t stream) {
    const float* qm  = (const float*)d_in[0];
    const float* km  = (const float*)d_in[1];
    const float* vm  = (const float*)d_in[2];
    const float* wts = (const float*)d_in[3];
    // d_in[4] attention_mask, d_in[5] position_ids: reconstructed analytically

    float* outp = (float*)d_out;
    unsigned short* wsp = (unsigned short*)d_ws;

    hipMemsetAsync(d_out, 0, (size_t)out_size * sizeof(float), stream);

    mixprep_kernel<<<BB * TT, 256, 0, stream>>>(qm, km, vm, wts, wsp);

    flashattn_kernel<8, 256><<<BB * 8 * (TT / 64), 128, 0, stream>>>(
        wsp + QOFF0, wsp + KOFF0, wsp + VOFF0, outp);
    flashattn_kernel<16, 128><<<BB * 16 * (TT / 64), 128, 0, stream>>>(
        wsp + QOFF1, wsp + KOFF1, wsp + VOFF1, outp);
}
